// Transformer_72018011619871
// MI455X (gfx1250) — compile-verified, matrix operand order checked by reading, not run on hardware
//
#include <hip/hip_runtime.h>
#include <hip/hip_bf16.h>

// ---------------------------------------------------------------------------
// CDNA5 (gfx1250) transformer encoder forward.
// Matmuls: v_wmma_f32_16x16x32_bf16 (wave32). Activations between GEMMs kept
// bf16 in global; LN/residual state kept f32. Flash attention uses a WMMA
// ones-column trick for softmax row sums (no cross-lane sum shuffles).
// ---------------------------------------------------------------------------

typedef __attribute__((ext_vector_type(16))) __bf16 v16bf;
typedef __attribute__((ext_vector_type(8)))  __bf16 v8bf;
typedef __attribute__((ext_vector_type(8)))  float  v8f;
typedef __attribute__((ext_vector_type(4)))  float  v4f;

union Frag16 {           // one 16x32 bf16 operand fragment (8 VGPRs)
    v16bf v;
    v8bf  h[2];
};

#define WMMA_BF16(Af, Bf, Cacc) \
    __builtin_amdgcn_wmma_f32_16x16x32_bf16(false, (Af), false, (Bf), (short)0, (Cacc), false, false)

__device__ __forceinline__ unsigned pack2(__bf16 a, __bf16 b) {
    union { __bf16 h[2]; unsigned u; } x;
    x.h[0] = a; x.h[1] = b;
    return x.u;
}

// ---------------------------------------------------------------------------
// GEMM:  C[M,N] = act(A[M,K] @ W[K,N] + bias[N])
//   A: f32 or bf16 (template), W/bias: f32, C: f32 or bf16 (template).
// Block tile 128x256, K-step 32, double-buffered LDS, 256 threads = 8 waves
// (2x4 wave grid), each wave computes 64x64 via 4x4 WMMA tiles
// (16 ds_load_b128 feed 16 WMMAs per K-step).
// ---------------------------------------------------------------------------
#define APITCH 40   // bf16 elems per LDS row (32 + 8 pad) -> 80B rows, 16B aligned
#define BPITCH 40

template<bool A_BF16, bool OUT_BF16, bool RELU>
__global__ __launch_bounds__(256) void gemm_wmma(
    const void* __restrict__ Ap, const float* __restrict__ Bw,
    const float* __restrict__ bias, void* __restrict__ Cp,
    int M, int N, int K)
{
    __shared__ __align__(16) __bf16 As[2][128 * APITCH];  // [row][k]
    __shared__ __align__(16) __bf16 Bs[2][256 * BPITCH];  // [n][k] (transposed)

    const int tid  = threadIdx.x;
    const int wid  = tid >> 5;
    const int lane = tid & 31;
    const int bm   = blockIdx.y * 128;
    const int bn   = blockIdx.x * 256;
    const int waveM = (wid >> 2) * 64;     // 0 or 64
    const int waveN = (wid & 3) * 64;      // 0,64,128,192

    // staging assignments
    const int s_row = tid >> 1;            // A: row 0..127
    const int s_kk  = (tid & 1) * 16;      // A: k 0 or 16
    const int b_n   = tid;                 // B: n column (lane-coalesced global reads)

    const float*  Af = (const float*)Ap;
    const __bf16* Ab = (const __bf16*)Ap;

    v8f acc[4][4];
#pragma unroll
    for (int mt = 0; mt < 4; ++mt)
#pragma unroll
        for (int nt = 0; nt < 4; ++nt)
#pragma unroll
            for (int j = 0; j < 8; ++j) acc[mt][nt][j] = 0.0f;

    const int arow = lane & 15;
    const int asel = (lane >> 4) * 8;      // k sub-offset: 0 or 8

    // raw staging registers
    v4f  ar[4];
    v8bf arb[2];
    float br[32];

    auto gload = [&](int k0) {
        if (A_BF16) {
            const __bf16* src = Ab + (size_t)(bm + s_row) * K + k0 + s_kk;
            arb[0] = *(const v8bf*)(src);
            arb[1] = *(const v8bf*)(src + 8);
        } else {
            const float* src = Af + (size_t)(bm + s_row) * K + k0 + s_kk;
            ar[0] = *(const v4f*)(src);
            ar[1] = *(const v4f*)(src + 4);
            ar[2] = *(const v4f*)(src + 8);
            ar[3] = *(const v4f*)(src + 12);
        }
#pragma unroll
        for (int i = 0; i < 32; ++i)
            br[i] = Bw[(size_t)(k0 + i) * N + bn + b_n];
    };

    auto sstore = [&](int buf) {
        __bf16* da = &As[buf][s_row * APITCH + s_kk];
        if (A_BF16) {
            *(v8bf*)(da)     = arb[0];
            *(v8bf*)(da + 8) = arb[1];
        } else {
            v8bf t0, t1;
#pragma unroll
            for (int i = 0; i < 4; ++i) {
                t0[i]     = (__bf16)ar[0][i];
                t0[i + 4] = (__bf16)ar[1][i];
                t1[i]     = (__bf16)ar[2][i];
                t1[i + 4] = (__bf16)ar[3][i];
            }
            *(v8bf*)(da)     = t0;
            *(v8bf*)(da + 8) = t1;
        }
        __bf16* db = &Bs[buf][b_n * BPITCH];
#pragma unroll
        for (int c = 0; c < 4; ++c) {
            v8bf tb;
#pragma unroll
            for (int i = 0; i < 8; ++i) tb[i] = (__bf16)br[c * 8 + i];
            *(v8bf*)(db + c * 8) = tb;
        }
    };

    // prologue: stage tile 0
    gload(0);
    sstore(0);
    __syncthreads();

    int cur = 0;
    for (int k0 = 0; k0 < K; k0 += 32) {
        const bool more = (k0 + 32) < K;
        if (more) gload(k0 + 32);                      // hide global latency
        if (k0 + 64 < K)                               // gfx1250 prefetch path
            __builtin_prefetch(&Bw[(size_t)(k0 + 64) * N + bn + b_n], 0, 1);

        // fragment loads from current buffer
        Frag16 af[4], bf_[4];
#pragma unroll
        for (int mt = 0; mt < 4; ++mt) {
            const __bf16* p = &As[cur][(waveM + mt * 16 + arow) * APITCH + asel];
            af[mt].h[0] = *(const v8bf*)(p);
            af[mt].h[1] = *(const v8bf*)(p + 16);
        }
#pragma unroll
        for (int nt = 0; nt < 4; ++nt) {
            const __bf16* p = &Bs[cur][(waveN + nt * 16 + arow) * BPITCH + asel];
            bf_[nt].h[0] = *(const v8bf*)(p);
            bf_[nt].h[1] = *(const v8bf*)(p + 16);
        }
#pragma unroll
        for (int mt = 0; mt < 4; ++mt)
#pragma unroll
            for (int nt = 0; nt < 4; ++nt)
                acc[mt][nt] = WMMA_BF16(af[mt].v, bf_[nt].v, acc[mt][nt]);

        if (more) sstore(cur ^ 1);                     // fill next buffer
        __syncthreads();
        cur ^= 1;
    }

    // epilogue: C-layout lane=(col, rowhalf), VGPR j = row j (+8)
    const int col0   = lane & 15;
    const int rowadd = (lane >> 4) * 8;
#pragma unroll
    for (int mt = 0; mt < 4; ++mt)
#pragma unroll
        for (int nt = 0; nt < 4; ++nt) {
            const int gcol = bn + waveN + nt * 16 + col0;
            const float bv = bias[gcol];
#pragma unroll
            for (int j = 0; j < 8; ++j) {
                const int grow = bm + waveM + mt * 16 + rowadd + j;
                float v = acc[mt][nt][j] + bv;
                if (RELU) v = fmaxf(v, 0.0f);
                if (OUT_BF16) ((__bf16*)Cp)[(size_t)grow * N + gcol] = (__bf16)v;
                else          ((float*) Cp)[(size_t)grow * N + gcol] = v;
            }
        }
}

// ---------------------------------------------------------------------------
// Flash attention. qkv: [B*S, 1536] bf16; per head h: q@h*192, k@+64, v@+128.
// grid (S/128, B*H), block 256 (8 waves). Wave w handles 16 q rows; the
// whole block shares one K/V staging per 64-key block (halves L2 traffic).
// Row sums of P computed by WMMA against an all-ones fragment.
// Output bf16 [B*S, 512].
// ---------------------------------------------------------------------------
__global__ __launch_bounds__(256) void flash_attn(
    const __bf16* __restrict__ qkv, __bf16* __restrict__ out)
{
    __shared__ __align__(16) __bf16 Ks[64 * 80];       // [key][hd]  pitch 80
    __shared__ __align__(16) __bf16 Vt[64 * 80];       // [hd][key]  pitch 80
    __shared__ __align__(16) __bf16 Pl[8][16 * 80];    // per-wave P [row][key]

    const int tid  = threadIdx.x;
    const int wid  = tid >> 5;
    const int lane = tid & 31;
    const int bh   = blockIdx.y;
    const int b    = bh >> 3;
    const int h    = bh & 7;
    const int q0   = blockIdx.x * 128 + wid * 16;

    const size_t rowbase = (size_t)b * 1024 * 1536;
    const int qcol = h * 192;

    const int r    = lane & 15;
    const int koff = (lane >> 4) * 8;

    // ---- Q fragments (prescaled by 1/sqrt(64) = 0.125, exact in bf16) ----
    Frag16 qf[2];
    {
        const __bf16* qrow = qkv + rowbase + (size_t)(q0 + r) * 1536 + qcol;
#pragma unroll
        for (int kk = 0; kk < 2; ++kk) {
#pragma unroll
            for (int c = 0; c < 2; ++c) {
                v8bf t = *(const v8bf*)(qrow + kk * 32 + koff + c * 16);
#pragma unroll
                for (int i = 0; i < 8; ++i) t[i] = (__bf16)((float)t[i] * 0.125f);
                qf[kk].h[c] = t;
            }
        }
    }

    // all-ones B fragment (registers only) for WMMA row sums of P
    Frag16 onesf;
#pragma unroll
    for (int i = 0; i < 8; ++i) { onesf.h[0][i] = (__bf16)1.0f; onesf.h[1][i] = (__bf16)1.0f; }

    float m[8];
    v8f lsum;            // running softmax denominators (one per row j)
    v8f accO[4];
#pragma unroll
    for (int j = 0; j < 8; ++j) { m[j] = -1e30f; lsum[j] = 0.0f; }
#pragma unroll
    for (int c = 0; c < 4; ++c)
#pragma unroll
        for (int j = 0; j < 8; ++j) accO[c][j] = 0.0f;

    // staging assignments (256 threads, 64 keys x 64 hd per tile)
    const int s_key = tid >> 2;            // K: key 0..63
    const int s_d0  = (tid & 3) * 16;      // K: hd 0,16,32,48
    const int v_kp  = tid & 31;            // V: key pair (2*kp, 2*kp+1)
    const int v_d0  = (tid >> 5) * 8;      // V: hd 0..56

    for (int kb = 0; kb < 16; ++kb) {      // 16 blocks of 64 keys
        const int kbase = kb * 64;
        // ---- stage K (row-major copy) and V (transposed, pair-packed) ----
        {
            const __bf16* sk = qkv + rowbase + (size_t)(kbase + s_key) * 1536 + qcol + 64 + s_d0;
            __bf16* dk = &Ks[s_key * 80 + s_d0];
            *(v8bf*)(dk)     = *(const v8bf*)(sk);
            *(v8bf*)(dk + 8) = *(const v8bf*)(sk + 8);

            const __bf16* sv0 = qkv + rowbase + (size_t)(kbase + 2 * v_kp) * 1536 + qcol + 128 + v_d0;
            v8bf a0 = *(const v8bf*)(sv0);
            v8bf b0 = *(const v8bf*)(sv0 + 1536);
#pragma unroll
            for (int i = 0; i < 8; ++i)
                *(unsigned*)&Vt[(v_d0 + i) * 80 + 2 * v_kp] = pack2(a0[i], b0[i]);

            if (kb < 15)
                __builtin_prefetch(qkv + rowbase + (size_t)(kbase + 64 + s_key) * 1536 + qcol + 64, 0, 1);
        }
        __syncthreads();

        // ---- logits: 4 n-tiles of 16 keys, accumulate over hd (2 steps) ----
        v8f sacc[4];
#pragma unroll
        for (int nt = 0; nt < 4; ++nt)
#pragma unroll
            for (int j = 0; j < 8; ++j) sacc[nt][j] = 0.0f;
#pragma unroll
        for (int nt = 0; nt < 4; ++nt) {
#pragma unroll
            for (int kk = 0; kk < 2; ++kk) {
                Frag16 kf;
                const __bf16* p = &Ks[(nt * 16 + r) * 80 + kk * 32 + koff];
                kf.h[0] = *(const v8bf*)(p);
                kf.h[1] = *(const v8bf*)(p + 16);
                sacc[nt] = WMMA_BF16(qf[kk].v, kf.v, sacc[nt]);
            }
        }

        // ---- online softmax: max via shfl (within 16-lane halves) ----
        float pn[4][8];
#pragma unroll
        for (int j = 0; j < 8; ++j) {
            float mx = fmaxf(fmaxf(sacc[0][j], sacc[1][j]),
                             fmaxf(sacc[2][j], sacc[3][j]));
#pragma unroll
            for (int off = 1; off < 16; off <<= 1)
                mx = fmaxf(mx, __shfl_xor(mx, off, 32));
            const float mnew  = fmaxf(m[j], mx);
            const float alpha = __expf(m[j] - mnew);
            m[j] = mnew;
#pragma unroll
            for (int nt = 0; nt < 4; ++nt) pn[nt][j] = __expf(sacc[nt][j] - mnew);
            lsum[j] *= alpha;
#pragma unroll
            for (int c = 0; c < 4; ++c) accO[c][j] *= alpha;
        }

        // ---- C-layout -> A-layout for P via wave-private LDS ----
        {
            __bf16* pw = Pl[wid];
            const int rowadd = (lane >> 4) * 8;
#pragma unroll
            for (int nt = 0; nt < 4; ++nt)
#pragma unroll
                for (int j = 0; j < 8; ++j)
                    pw[(j + rowadd) * 80 + nt * 16 + (lane & 15)] = (__bf16)pn[nt][j];

            Frag16 pf[2];
#pragma unroll
            for (int kk = 0; kk < 2; ++kk) {
                const __bf16* pp = &pw[r * 80 + kk * 32 + koff];
                pf[kk].h[0] = *(const v8bf*)(pp);
                pf[kk].h[1] = *(const v8bf*)(pp + 16);
            }

            // softmax denominators: row sums of P via WMMA against ones
            lsum = WMMA_BF16(pf[0].v, onesf.v, lsum);
            lsum = WMMA_BF16(pf[1].v, onesf.v, lsum);

            // ---- accO += P @ V  (4 hd tiles of 16, 2 k-steps each) ----
#pragma unroll
            for (int c = 0; c < 4; ++c) {
#pragma unroll
                for (int kk = 0; kk < 2; ++kk) {
                    Frag16 vf;
                    const __bf16* vp = &Vt[(c * 16 + r) * 80 + kk * 32 + koff];
                    vf.h[0] = *(const v8bf*)(vp);
                    vf.h[1] = *(const v8bf*)(vp + 16);
                    accO[c] = WMMA_BF16(pf[kk].v, vf.v, accO[c]);
                }
            }
        }
        __syncthreads();
    }

    // ---- normalize and write bf16 output [B*S, 512] ----
    const int rowadd = (lane >> 4) * 8;
#pragma unroll
    for (int j = 0; j < 8; ++j) {
        const float inv = 1.0f / lsum[j];
        const int row = q0 + j + rowadd;
#pragma unroll
        for (int c = 0; c < 4; ++c) {
            const int col = h * 64 + c * 16 + (lane & 15);
            out[((size_t)b * 1024 + row) * 512 + col] = (__bf16)(accO[c][j] * inv);
        }
    }
}

// ---------------------------------------------------------------------------
// Small helper kernels (negligible FLOPs)
// ---------------------------------------------------------------------------
__device__ __forceinline__ float block_sum256(float v, float* red)
{
#pragma unroll
    for (int off = 16; off; off >>= 1) v += __shfl_xor(v, off, 32);
    if ((threadIdx.x & 31) == 0) red[threadIdx.x >> 5] = v;
    __syncthreads();
    float t = 0.0f;
#pragma unroll
    for (int i = 0; i < 8; ++i) t += red[i];
    __syncthreads();
    return t;
}

// h[row] = LN(h[row] + r[row]) * g + b   (row = 512)
__global__ __launch_bounds__(256) void add_ln(
    float* __restrict__ h, const float* __restrict__ rsd,
    const float* __restrict__ g, const float* __restrict__ bta)
{
    __shared__ float red[8];
    const size_t base = (size_t)blockIdx.x * 512;
    const int t = threadIdx.x;
    float x0 = h[base + t]       + rsd[base + t];
    float x1 = h[base + 256 + t] + rsd[base + 256 + t];
    const float mu = block_sum256(x0 + x1, red) * (1.0f / 512.0f);
    const float d0 = x0 - mu, d1 = x1 - mu;
    const float var = block_sum256(d0 * d0 + d1 * d1, red) * (1.0f / 512.0f);
    const float rs = rsqrtf(var + 1e-5f);
    h[base + t]       = d0 * rs * g[t]       + bta[t];
    h[base + 256 + t] = d1 * rs * g[256 + t] + bta[256 + t];
}

// h = x @ input_W + input_b + pe[positions]     (K = 24)
__global__ __launch_bounds__(256) void input_proj(
    const float* __restrict__ x, const int* __restrict__ pos,
    const float* __restrict__ pe, const float* __restrict__ W,
    const float* __restrict__ bias, float* __restrict__ h)
{
    __shared__ float xs[24];
    const int row = blockIdx.y;
    const int d   = blockIdx.x * 256 + threadIdx.x;
    if (threadIdx.x < 24) xs[threadIdx.x] = x[(size_t)row * 24 + threadIdx.x];
    __syncthreads();
    const int p = pos[row];
    float acc = bias[d] + pe[(size_t)p * 512 + d];
#pragma unroll
    for (int k = 0; k < 24; ++k) acc += xs[k] * W[k * 512 + d];
    h[(size_t)row * 512 + d] = acc;
}

__global__ __launch_bounds__(256) void mean_pool(
    const float* __restrict__ h, float* __restrict__ pooled)
{
    const int b = blockIdx.x;
    for (int d = threadIdx.x; d < 512; d += 256) {
        float s = 0.0f;
        for (int srow = 0; srow < 1024; ++srow)
            s += h[((size_t)b * 1024 + srow) * 512 + d];
        pooled[b * 512 + d] = s * (1.0f / 1024.0f);
    }
}

// p1[8,512] = pooled @ out1_W + out1_b
__global__ __launch_bounds__(256) void head1(
    const float* __restrict__ p, const float* __restrict__ W,
    const float* __restrict__ b, float* __restrict__ o)
{
    const int idx = blockIdx.x * 256 + threadIdx.x;   // < 4096
    const int row = idx >> 9, col = idx & 511;
    float acc = b[col];
    for (int k = 0; k < 512; ++k) acc += p[row * 512 + k] * W[k * 512 + col];
    o[idx] = acc;
}

// p2 = relu(LN(p1) * g + b)   rows of 512
__global__ __launch_bounds__(256) void ln_relu(
    const float* __restrict__ p, const float* __restrict__ g,
    const float* __restrict__ bta, float* __restrict__ o)
{
    __shared__ float red[8];
    const size_t base = (size_t)blockIdx.x * 512;
    const int t = threadIdx.x;
    float x0 = p[base + t];
    float x1 = p[base + 256 + t];
    const float mu = block_sum256(x0 + x1, red) * (1.0f / 512.0f);
    const float d0 = x0 - mu, d1 = x1 - mu;
    const float var = block_sum256(d0 * d0 + d1 * d1, red) * (1.0f / 512.0f);
    const float rs = rsqrtf(var + 1e-5f);
    o[base + t]       = fmaxf(d0 * rs * g[t]       + bta[t],       0.0f);
    o[base + 256 + t] = fmaxf(d1 * rs * g[256 + t] + bta[256 + t], 0.0f);
}

// out[b] = p2[b] . out2_W + out2_b
__global__ __launch_bounds__(256) void head2(
    const float* __restrict__ p, const float* __restrict__ W,
    const float* __restrict__ b, float* __restrict__ o)
{
    const int wid = threadIdx.x >> 5, lane = threadIdx.x & 31;
    float acc = 0.0f;
    for (int d = lane; d < 512; d += 32) acc += p[wid * 512 + d] * W[d];
#pragma unroll
    for (int off = 16; off; off >>= 1) acc += __shfl_xor(acc, off, 32);
    if (lane == 0) o[wid] = acc + b[0];
}

// ---------------------------------------------------------------------------
extern "C" void kernel_launch(void* const* d_in, const int* in_sizes, int n_in,
                              void* d_out, int out_size, void* d_ws, size_t ws_size,
                              hipStream_t stream)
{
    const float* x    = (const float*)d_in[0];
    const int*   pos  = (const int*)  d_in[1];
    const float* pe   = (const float*)d_in[2];
    const float* inW  = (const float*)d_in[3];
    const float* inB  = (const float*)d_in[4];
    const float* qkvW = (const float*)d_in[5];
    const float* qkvB = (const float*)d_in[6];
    const float* oW   = (const float*)d_in[7];
    const float* oB   = (const float*)d_in[8];
    const float* f1W  = (const float*)d_in[9];
    const float* f1B  = (const float*)d_in[10];
    const float* f2W  = (const float*)d_in[11];
    const float* f2B  = (const float*)d_in[12];
    const float* ln1g = (const float*)d_in[13];
    const float* ln1b = (const float*)d_in[14];
    const float* ln2g = (const float*)d_in[15];
    const float* ln2b = (const float*)d_in[16];
    const float* o1W  = (const float*)d_in[17];
    const float* o1B  = (const float*)d_in[18];
    const float* olng = (const float*)d_in[19];
    const float* olnb = (const float*)d_in[20];
    const float* o2W  = (const float*)d_in[21];
    const float* o2B  = (const float*)d_in[22];
    float* out = (float*)d_out;

    // workspace: h f32 16MB | qkv bf16 24MB (reused as ff1o) | attn bf16 8MB |
    //            buf2 f32 16MB | heads
    char* ws = (char*)d_ws;
    float*  h      = (float*)(ws);
    __bf16* qkv    = (__bf16*)(ws + (16ull << 20));
    __bf16* attnb  = (__bf16*)(ws + (40ull << 20));
    float*  buf2   = (float*)(ws + (48ull << 20));
    float*  pooled = (float*)(ws + (64ull << 20));
    float*  p1     = pooled + 4096;
    float*  p2     = p1 + 4096;
    __bf16* ff1o   = qkv;

    const int M = 8192;     // B*S

    input_proj<<<dim3(2, M), 256, 0, stream>>>(x, pos, pe, inW, inB, h);

    for (int l = 0; l < 6; ++l) {
        gemm_wmma<false, true, false><<<dim3(1536 / 256, M / 128), 256, 0, stream>>>(
            h, qkvW + (size_t)l * 512 * 1536, qkvB + l * 1536, qkv, M, 1536, 512);
        flash_attn<<<dim3(8, 64), 256, 0, stream>>>(qkv, attnb);
        gemm_wmma<true, false, false><<<dim3(512 / 256, M / 128), 256, 0, stream>>>(
            attnb, oW + (size_t)l * 512 * 512, oB + l * 512, buf2, M, 512, 512);
        add_ln<<<M, 256, 0, stream>>>(h, buf2, ln1g + l * 512, ln1b + l * 512);
        gemm_wmma<false, true, true><<<dim3(1024 / 256, M / 128), 256, 0, stream>>>(
            h, f1W + (size_t)l * 512 * 1024, f1B + l * 1024, ff1o, M, 1024, 512);
        gemm_wmma<true, false, false><<<dim3(512 / 256, M / 128), 256, 0, stream>>>(
            ff1o, f2W + (size_t)l * 1024 * 512, f2B + l * 512, buf2, M, 512, 1024);
        add_ln<<<M, 256, 0, stream>>>(h, buf2, ln2g + l * 512, ln2b + l * 512);
    }

    mean_pool<<<8, 256, 0, stream>>>(h, pooled);
    head1<<<16, 256, 0, stream>>>(pooled, o1W, o1B, p1);
    ln_relu<<<8, 256, 0, stream>>>(p1, olng, olnb, p2);
    head2<<<1, 256, 0, stream>>>(p2, o2W, o2B, out);
}
